// TreeLSTMCell_22247930593470
// MI455X (gfx1250) — compile-verified, matrix operand order checked by reading, not run on hardware
//
#include <hip/hip_runtime.h>
#include <stdint.h>

// ---- CDNA5 (gfx1250) wave32 WMMA types ----
typedef __attribute__((ext_vector_type(16))) __bf16 v16bf;
typedef __attribute__((ext_vector_type(4)))  __bf16 v4bf;
typedef __attribute__((ext_vector_type(8)))  float  v8f;
typedef __attribute__((ext_vector_type(4)))  float  v4f;

#define BATCH 8192
#define DIM   512     // INPUT_DIM == HIDDEN_DIM == 512
#define NCH   8
#define BM    64      // batch rows per block
#define BN    64      // hidden cols per block
#define BK    32      // K chunk == WMMA K for bf16

// Convert 4 floats -> 4 bf16 with the packed hardware converter and store as one b64.
static __device__ __forceinline__ void st4(__bf16* d, v4f v) {
  *(v4bf*)d = __builtin_convertvector(v, v4bf);
}

// Load a 16x32 bf16 fragment from an LDS tile stored row-major with stride BK.
// Lane L in [0,15]: row rowOff+L, K=0..15 ; lanes 16..31: same rows, K=16..31.
// Same packing serves B fragments when the staged tile is W (row = output index n).
static __device__ __forceinline__ v16bf ldfrag(const __bf16* base, int rowOff) {
  const int lane = threadIdx.x & 31;
  const __bf16* p = base + (rowOff + (lane & 15)) * BK + ((lane >> 4) << 4);
  return *(const v16bf*)p;
}

static __device__ __forceinline__ float sigm(float x) {
  return 1.0f / (1.0f + __expf(-x));
}
// tanh(x) = 1 - 2/(exp(2x)+1); saturates correctly at +/-inf, uses v_exp_f32.
static __device__ __forceinline__ float fast_tanh(float x) {
  return 1.0f - 2.0f / (__expf(2.0f * x) + 1.0f);
}

#define WMMA_BF16(A, B, C) \
  __builtin_amdgcn_wmma_f32_16x16x32_bf16(false, (A), false, (B), (short)0, (C), false, false)

__global__ __launch_bounds__(512)
void treelstm_fused_kernel(const float* __restrict__ x,
                           const float* __restrict__ child_h,
                           const float* __restrict__ child_c,
                           const float* __restrict__ W_i, const float* __restrict__ b_i, const float* __restrict__ U_i,
                           const float* __restrict__ W_f, const float* __restrict__ b_f, const float* __restrict__ U_f,
                           const float* __restrict__ W_o, const float* __restrict__ b_o, const float* __restrict__ U_o,
                           const float* __restrict__ W_u, const float* __restrict__ b_u, const float* __restrict__ U_u,
                           float* __restrict__ out)
{
  __shared__ __bf16 As [BM][BK];          // x chunk (bf16)
  __shared__ __bf16 Sum[BM][BK];          // on-the-fly child-sum chunk
  __shared__ __bf16 Hs [NCH][BM][BK];     // per-child h chunks
  __shared__ __bf16 Bs [4][BN][BK];       // 4 weight tiles (rows = output n, cols = k)

  const int cTile = blockIdx.x * BN;      // hidden-column tile
  const int rTile = blockIdx.y * BM;      // batch-row tile
  const int t  = threadIdx.x;             // 512 threads = 16 waves
  const int w  = t >> 5;
  const int mi = w >> 2;                  // 0..3  (m sub-tile)
  const int ni = w & 3;                   // 0..3  (n sub-tile)

  const int srow = t >> 3;                // staging: 0..63
  const int sc4  = (t & 7) << 2;          // staging: 0,4,...,28

  v8f acc_i = {}, acc_o = {}, acc_u = {}, acc_wf = {};
  v8f acc_f[NCH];
  #pragma unroll
  for (int k = 0; k < NCH; ++k) acc_f[k] = (v8f){};

  // ---------------- Phase 1: x @ {W_i,W_o,W_u,W_f}^T ----------------
  const float* Wg[4] = { W_i, W_o, W_u, W_f };
  for (int kc = 0; kc < DIM; kc += BK) {
    st4(&As[srow][sc4], *(const v4f*)(x + (size_t)(rTile + srow) * DIM + kc + sc4));
    #pragma unroll
    for (int g = 0; g < 4; ++g) {
      st4(&Bs[g][srow][sc4], *(const v4f*)(Wg[g] + (size_t)(cTile + srow) * DIM + kc + sc4));
    }
    __syncthreads();

    const v16bf a = ldfrag(&As[0][0], mi * 16);
    acc_i  = WMMA_BF16(a, ldfrag(&Bs[0][0][0], ni * 16), acc_i);
    acc_o  = WMMA_BF16(a, ldfrag(&Bs[1][0][0], ni * 16), acc_o);
    acc_u  = WMMA_BF16(a, ldfrag(&Bs[2][0][0], ni * 16), acc_u);
    acc_wf = WMMA_BF16(a, ldfrag(&Bs[3][0][0], ni * 16), acc_wf);
    __syncthreads();
  }

  // ------- Phase 2: h_sum @ {U_i,U_o,U_u}^T and child_h_k @ U_f^T -------
  const float* Ug[4] = { U_i, U_o, U_u, U_f };
  for (int kc = 0; kc < DIM; kc += BK) {
    v4f s = {};
    #pragma unroll
    for (int k = 0; k < NCH; ++k) {
      const v4f v = *(const v4f*)(child_h + ((size_t)k * BATCH + rTile + srow) * DIM + kc + sc4);
      s += v;
      st4(&Hs[k][srow][sc4], v);
    }
    st4(&Sum[srow][sc4], s);
    #pragma unroll
    for (int g = 0; g < 4; ++g) {
      st4(&Bs[g][srow][sc4], *(const v4f*)(Ug[g] + (size_t)(cTile + srow) * DIM + kc + sc4));
    }
    __syncthreads();

    const v16bf sa = ldfrag(&Sum[0][0], mi * 16);
    acc_i = WMMA_BF16(sa, ldfrag(&Bs[0][0][0], ni * 16), acc_i);
    acc_o = WMMA_BF16(sa, ldfrag(&Bs[1][0][0], ni * 16), acc_o);
    acc_u = WMMA_BF16(sa, ldfrag(&Bs[2][0][0], ni * 16), acc_u);

    const v16bf bfF = ldfrag(&Bs[3][0][0], ni * 16);   // U_f fragment, reused by all 8 children
    #pragma unroll
    for (int k = 0; k < NCH; ++k) {
      const v16bf ha = ldfrag(&Hs[k][0][0], mi * 16);
      acc_f[k] = WMMA_BF16(ha, bfF, acc_f[k]);
    }
    __syncthreads();
  }

  // ---------------- Epilogue: gates, per-child forget, h & c ----------------
  // C/D fragment mapping (16x16 f32, 8 VGPRs): reg r, lanes 0-15 -> row r,
  // lanes 16-31 -> row r+8; column = lane & 15.
  const int lane = threadIdx.x & 31;
  const int col  = cTile + ni * 16 + (lane & 15);
  const int rb   = rTile + mi * 16 + ((lane >> 4) << 3);

  const float bi = b_i[col], bo = b_o[col], bu = b_u[col], bf = b_f[col];

  float* __restrict__ out_h = out;
  float* __restrict__ out_c = out + (size_t)BATCH * DIM;

  #pragma unroll
  for (int r = 0; r < 8; ++r) {
    const int row = rb + r;
    const float iv = sigm(acc_i[r] + bi);
    const float ov = sigm(acc_o[r] + bo);
    const float uv = fast_tanh(acc_u[r] + bu);
    const float wf = acc_wf[r] + bf;

    float fcv = 0.f;
    #pragma unroll
    for (int k = 0; k < NCH; ++k) {
      const float cc = child_c[((size_t)k * BATCH + row) * DIM + col];
      fcv += sigm(wf + acc_f[k][r]) * cc;
    }
    const float cv = iv * uv + fcv;
    out_c[(size_t)row * DIM + col] = cv;
    out_h[(size_t)row * DIM + col] = ov * fast_tanh(cv);
  }
}

extern "C" void kernel_launch(void* const* d_in, const int* in_sizes, int n_in,
                              void* d_out, int out_size, void* d_ws, size_t ws_size,
                              hipStream_t stream) {
  (void)in_sizes; (void)n_in; (void)d_ws; (void)ws_size; (void)out_size;
  const float* x       = (const float*)d_in[0];
  const float* child_h = (const float*)d_in[1];
  const float* child_c = (const float*)d_in[2];
  const float* W_i = (const float*)d_in[3];
  const float* b_i = (const float*)d_in[4];
  const float* U_i = (const float*)d_in[5];
  const float* W_f = (const float*)d_in[6];
  const float* b_f = (const float*)d_in[7];
  const float* U_f = (const float*)d_in[8];
  const float* W_o = (const float*)d_in[9];
  const float* b_o = (const float*)d_in[10];
  const float* U_o = (const float*)d_in[11];
  const float* W_u = (const float*)d_in[12];
  const float* b_u = (const float*)d_in[13];
  const float* U_u = (const float*)d_in[14];

  dim3 grid(DIM / BN, BATCH / BM);   // 8 x 128 tiles
  dim3 block(512);                   // 16 wave32 waves
  treelstm_fused_kernel<<<grid, block, 0, stream>>>(
      x, child_h, child_c,
      W_i, b_i, U_i, W_f, b_f, U_f, W_o, b_o, U_o, W_u, b_u, U_u,
      (float*)d_out);
}
